// RecursiveTreeComputation_58085137711397
// MI455X (gfx1250) — matrix operand order repaired and checked by reading, not compiled, not run on hardware
//
#include <hip/hip_runtime.h>
#include <math.h>

typedef __attribute__((ext_vector_type(16))) _Float16 v16h;
typedef __attribute__((ext_vector_type(8)))  float    v8f;

#define TANH_INV_SCALAR 0.1f

// ---------------------------------------------------------------------------
// Kernel 1: scores[i] = 1 + tanh(features[i,:64] @ W + b) / 10
// One wave32 handles 16 rows via V_WMMA_F32_16X16X32_F16 (two K=32 chunks).
// A fragment: lane<16 -> row m, K {0..7,16..23}; lane>=16 -> row m, K {8..15,24..31}
// B fragment: column = lane&15; only column 0 carries W, rest zero.
// D column 0: lane 0 holds rows m=0..7 in c[0..7], lane 16 holds rows 8..15.
// ---------------------------------------------------------------------------
__global__ void __launch_bounds__(256)
irene_scores_wmma(const float* __restrict__ feat,
                  const float* __restrict__ W,
                  const float* __restrict__ bias_p,
                  float* __restrict__ scores,
                  int n)
{
    const int lane = threadIdx.x & 31;
    const int m    = lane & 15;
    const int half = lane >> 4;
    const long wave = ((long)blockIdx.x * blockDim.x + threadIdx.x) >> 5;
    const long R0   = wave * 16;          // first row of this wave's tile
    const long row  = R0 + m;
    const float bias = bias_p[0];

    v8f acc = {};

#pragma unroll
    for (int ch = 0; ch < 2; ++ch) {
        const int lo = ch * 32 + half * 8;        // K run 0 (8 elems)
        const int hi = ch * 32 + 16 + half * 8;   // K run 1 (8 elems)

        // ---- A fragment (16x32 f16, ISA layout) ----
        v16h a = {};
        if (row < n) {
            const float* fr = feat + row * 64;
#pragma unroll
            for (int j = 0; j < 8; ++j) a[j]     = (_Float16)fr[lo + j];
#pragma unroll
            for (int j = 0; j < 8; ++j) a[8 + j] = (_Float16)fr[hi + j];
        }

        // ---- B fragment (32x16 f16): column 0 = W chunk, others 0 ----
        v16h bm = {};
        if (m == 0) {
#pragma unroll
            for (int j = 0; j < 8; ++j) bm[j]     = (_Float16)W[lo + j];
#pragma unroll
            for (int j = 0; j < 8; ++j) bm[8 + j] = (_Float16)W[hi + j];
        }

        acc = __builtin_amdgcn_wmma_f32_16x16x32_f16(
                  /*neg_a=*/false, a, /*neg_b=*/false, bm,
                  /*c_mod=*/(short)0, acc, /*reuse_a=*/false, /*reuse_b=*/false);
    }

    // Column N=0 of D lives in lanes 0 and 16.
    if (m == 0) {
#pragma unroll
        for (int r = 0; r < 8; ++r) {
            const long orow = R0 + half * 8 + r;
            if (orow < n)
                scores[orow] = 1.0f + tanhf(acc[r] + bias) * TANH_INV_SCALAR;
        }
    }
}

// ---------------------------------------------------------------------------
// Kernel 2: levels 20 -> 10. Block b owns the subtree rooted at node 1023+b.
// Copies leaf energies to out, then reduces sibling pairs level by level in
// LDS (ping-pong buffers), writing every intermediate node's energy.
// Tree: level d occupies [2^d - 1, 2^{d+1} - 1); parent(i) = (i-1)/2.
// ---------------------------------------------------------------------------
__global__ void __launch_bounds__(512)
irene_tree_lower(const float* __restrict__ leaf_energy,
                 const float* __restrict__ scores,
                 float* __restrict__ out)
{
    __shared__ float sm[2][1024];
    float* cur = sm[0];
    float* nxt = sm[1];

    const int b   = blockIdx.x;       // 0..1023 : subtree id at level 10
    const int tid = threadIdx.x;

    // Leaf level d=20: 1024 leaves per subtree, starting at leaf index b*1024.
    const long S20 = (1L << 20) - 1;
    for (int t = tid; t < 1024; t += 512) {
        const long li = (long)b * 1024 + t;
        const float v = leaf_energy[li];
        out[S20 + li] = v;            // leaves keep raw leaf_energy
        cur[t] = v;
    }
    __syncthreads();

    // d = 20 .. 11: produce level d-1 (written to out and kept in LDS)
    for (int d = 20; d >= 11; --d) {
        const int  pc = 1 << (d - 11);                          // parents this step
        const long cs = ((1L << d) - 1)       + ((long)b << (d - 10)); // child start
        const long ps = ((1L << (d - 1)) - 1) + ((long)b << (d - 11)); // parent start
        for (int t = tid; t < pc; t += 512) {
            const float e0 = cur[2 * t];
            const float e1 = cur[2 * t + 1];
            const float v  = scores[cs + 2 * t] * e0 + scores[cs + 2 * t + 1] * e1;
            out[ps + t] = v;          // internal nodes start at 0 -> assignment
            nxt[t] = v;
        }
        __syncthreads();
        float* tmp = cur; cur = nxt; nxt = tmp;
    }
}

// ---------------------------------------------------------------------------
// Kernel 3: levels 10 -> 0, single block (max 1024 children fit in LDS).
// ---------------------------------------------------------------------------
__global__ void __launch_bounds__(512)
irene_tree_upper(const float* __restrict__ scores,
                 float* __restrict__ out)
{
    __shared__ float sm[2][1024];
    float* cur = sm[0];
    float* nxt = sm[1];
    const int tid = threadIdx.x;

    // Load level-10 energies (indices 1023 .. 2046), produced by kernel 2.
    for (int t = tid; t < 1024; t += 512)
        cur[t] = out[1023 + t];
    __syncthreads();

    for (int d = 10; d >= 1; --d) {
        const int  pc = 1 << (d - 1);
        const long cs = (1L << d) - 1;
        const long ps = (1L << (d - 1)) - 1;
        for (int t = tid; t < pc; t += 512) {
            const float e0 = cur[2 * t];
            const float e1 = cur[2 * t + 1];
            const float v  = scores[cs + 2 * t] * e0 + scores[cs + 2 * t + 1] * e1;
            out[ps + t] = v;
            nxt[t] = v;
        }
        __syncthreads();
        float* tmp = cur; cur = nxt; nxt = tmp;
    }
}

// ---------------------------------------------------------------------------
extern "C" void kernel_launch(void* const* d_in, const int* in_sizes, int n_in,
                              void* d_out, int out_size, void* d_ws, size_t ws_size,
                              hipStream_t stream)
{
    const float* feat  = (const float*)d_in[0];   // [N, 64] f32
    const float* leafE = (const float*)d_in[1];   // [2^20]  f32
    const float* W     = (const float*)d_in[2];   // [64]    f32
    const float* bias  = (const float*)d_in[3];   // [1]     f32
    float*       out   = (float*)d_out;           // [N]     f32
    float*       scores = (float*)d_ws;           // [N]     f32 scratch (8 MB)

    const int n = in_sizes[0] / 64;               // N_NODES = 2^21 - 1

    // Kernel 1: WMMA matvec + tanh scoring. 16 rows per wave, 8 waves/block.
    const long waves  = ((long)n + 15) / 16;
    const int  blocks = (int)((waves * 32 + 255) / 256);
    irene_scores_wmma<<<blocks, 256, 0, stream>>>(feat, W, bias, scores, n);

    // Kernel 2: levels 20->10 (1024 subtrees, one per block).
    irene_tree_lower<<<1024, 512, 0, stream>>>(leafE, scores, out);

    // Kernel 3: levels 10->0 (single block).
    irene_tree_upper<<<1, 512, 0, stream>>>(scores, out);
}
